// SimpleSSM_62088047231045
// MI455X (gfx1250) — matrix operand order repaired and checked by reading, not compiled, not run on hardware
//
#include <hip/hip_runtime.h>
#include <hip/hip_bf16.h>

typedef __attribute__((ext_vector_type(16))) _Float16 v16h;
typedef __attribute__((ext_vector_type(8)))  float    v8f;
typedef __attribute__((ext_vector_type(4)))  int      v4i;

#define D_MODEL 1024
#define D_STATE 16
#define BATCH   8
#define SEQ     4096
#define MROWS   (BATCH * SEQ)   // 32768
#define KCOMB   2048

#if defined(__has_builtin)
#if __has_builtin(__builtin_amdgcn_global_load_async_to_lds_b128)
#define HAVE_ASYNC_LDS 1
#endif
#endif
#ifndef HAVE_ASYNC_LDS
#define HAVE_ASYNC_LDS 0
#endif

union FragH {
    v16h v;
    uint4 q[2];
    _Float16 e[16];
};

__device__ __forceinline__ v8f wmma_f16(v16h a, v16h b, v8f c) {
    // D = A(16x32,f16) * B(32x16,f16) + C(16x16,f32)
    return __builtin_amdgcn_wmma_f32_16x16x32_f16(
        /*neg_a=*/false, a, /*neg_b=*/false, b,
        /*c_mod=*/(short)0, c, /*reuse_a=*/false, /*reuse_b=*/false);
}

#if HAVE_ASYNC_LDS
typedef __attribute__((address_space(1))) v4i as1_v4i;   // global int4
typedef __attribute__((address_space(3))) v4i as3_v4i;   // LDS int4
// Per-lane 16B async copy global -> LDS (ASYNCcnt-tracked, no VGPR staging)
__device__ __forceinline__ void async_copy16(const void* gsrc, void* ldst) {
    __builtin_amdgcn_global_load_async_to_lds_b128(
        (as1_v4i*)gsrc, (as3_v4i*)ldst, 0, 0);
}
#define ASYNC_WAIT() asm volatile("s_wait_asynccnt 0x0" ::: "memory")
#else
#define ASYNC_WAIT() do {} while (0)
#endif

// fast numerically-stable softplus: max(x,0) + log(1 + exp(-|x|))
__device__ __forceinline__ float softplus_fast(float v) {
    return fmaxf(v, 0.0f) + __logf(1.0f + __expf(-fabsf(v)));
}

// ---------------------------------------------------------------------------
// Prep: Wcomb[n,k] = f16( k<1024 ? W_out[n,k] : W_skip[n,k-1024] )
// ---------------------------------------------------------------------------
__global__ void ssm_prep_wcomb(const float* __restrict__ W_out,
                               const float* __restrict__ W_skip,
                               _Float16* __restrict__ Wcomb) {
    long idx = (long)blockIdx.x * blockDim.x + threadIdx.x;
    if (idx >= (long)D_MODEL * KCOMB) return;
    long n = idx >> 11;          // / 2048
    long k = idx & (KCOMB - 1);
    float v = (k < D_MODEL) ? W_out[n * D_MODEL + k]
                            : W_skip[n * D_MODEL + (k - D_MODEL)];
    Wcomb[idx] = (_Float16)v;
}

// ---------------------------------------------------------------------------
// Prep: Wp[32,1024] = [W_dt ; W_x] f16, Ch[1024,32] = [C | 0] f16
// ---------------------------------------------------------------------------
__global__ void ssm_prep_small(const float* __restrict__ W_dt,
                               const float* __restrict__ W_x,
                               const float* __restrict__ C,
                               _Float16* __restrict__ Wp,
                               _Float16* __restrict__ Ch) {
    int idx = blockIdx.x * 256 + threadIdx.x;      // 32768 threads
    if (idx < 32 * D_MODEL) {
        int r = idx >> 10, k = idx & (D_MODEL - 1);
        float v = (r < D_STATE) ? W_dt[r * D_MODEL + k]
                                : W_x[(r - D_STATE) * D_MODEL + k];
        Wp[idx] = (_Float16)v;
    }
    if (idx < D_MODEL * 32) {
        int d = idx >> 5, j = idx & 31;
        float v = (j < D_STATE) ? C[d * D_STATE + j] : 0.0f;
        Ch[idx] = (_Float16)v;
    }
}

// ---------------------------------------------------------------------------
// Projection: P[M,32] = f16(x) @ Wp^T ; epilogue -> dt (softplus), bx
// One wave per 16-row tile, 2 WMMA column tiles, K loop 1024/32.
// ---------------------------------------------------------------------------
__global__ __launch_bounds__(256)
void ssm_proj(const float* __restrict__ x, const _Float16* __restrict__ Wp,
              const float* __restrict__ b_dt, const float* __restrict__ b_x,
              const float* __restrict__ Bp,
              float* __restrict__ dtb, float* __restrict__ bxb) {
    int wave = threadIdx.x >> 5;
    int lane = threadIdx.x & 31;
    long tile = (long)blockIdx.x * 8 + wave;   // 2048 tiles of 16 rows
    long m0 = tile * 16;
    int l15 = lane & 15, h = lane >> 4;

    v8f c0 = {}, c1 = {};
    const float* xrow = x + (m0 + l15) * D_MODEL;
    const char* wrow0 = (const char*)Wp + (size_t)l15 * (D_MODEL * 2);
    const char* wrow1 = (const char*)Wp + (size_t)(16 + l15) * (D_MODEL * 2);

    for (int ks = 0; ks < D_MODEL / 32; ++ks) {
        int k0 = ks * 32;
        // A fragment: convert f32 -> f16, interleaved half-lane K layout
        FragH a;
        const float4* pa = (const float4*)(xrow + k0 + 8 * h);
        float4 f0 = pa[0], f1 = pa[1];
        const float4* pb = (const float4*)(xrow + k0 + 16 + 8 * h);
        float4 f2 = pb[0], f3 = pb[1];
        a.e[0]=(_Float16)f0.x;  a.e[1]=(_Float16)f0.y;  a.e[2]=(_Float16)f0.z;  a.e[3]=(_Float16)f0.w;
        a.e[4]=(_Float16)f1.x;  a.e[5]=(_Float16)f1.y;  a.e[6]=(_Float16)f1.z;  a.e[7]=(_Float16)f1.w;
        a.e[8]=(_Float16)f2.x;  a.e[9]=(_Float16)f2.y;  a.e[10]=(_Float16)f2.z; a.e[11]=(_Float16)f2.w;
        a.e[12]=(_Float16)f3.x; a.e[13]=(_Float16)f3.y; a.e[14]=(_Float16)f3.z; a.e[15]=(_Float16)f3.w;

        // B fragments: interleaved half-lane K layout (lane = N column)
        FragH b0, b1;
        b0.q[0] = *(const uint4*)(wrow0 + k0 * 2 + 16 * h);
        b0.q[1] = *(const uint4*)(wrow0 + k0 * 2 + 32 + 16 * h);
        b1.q[0] = *(const uint4*)(wrow1 + k0 * 2 + 16 * h);
        b1.q[1] = *(const uint4*)(wrow1 + k0 * 2 + 32 + 16 * h);

        c0 = wmma_f16(a.v, b0.v, c0);   // -> dt pre-activation (cols 0..15)
        c1 = wmma_f16(a.v, b1.v, c1);   // -> x_proj          (cols 0..15)
    }

    int n = l15;
    float bdt = b_dt[n], bxc = b_x[n], bp = Bp[n];
    #pragma unroll
    for (int r = 0; r < 8; ++r) {
        long m = m0 + r + 8 * h;        // C/D layout: lane=N, vgpr r -> M=r / 8+r
        float dtv = softplus_fast(c0[r] + bdt);
        float xp  = c1[r] + bxc;
        dtb[m * D_STATE + n] = dtv;
        bxb[m * D_STATE + n] = dtv * bp * xp;
    }
}

// ---------------------------------------------------------------------------
// Sequential scan: h_t = exp(dt_t * A) h_{t-1} + bx_t, per batch.
// thread (i,j) owns A[i][j]; shuffle-reduce over j; double-buffered LDS h.
// Stores h as f16 padded [M,32] (upper 16 zero) for the WMMA y-kernel.
// ---------------------------------------------------------------------------
__global__ __launch_bounds__(256)
void ssm_scan(const float* __restrict__ dtb, const float* __restrict__ bxb,
              const float* __restrict__ A, _Float16* __restrict__ hh) {
    int b = blockIdx.x;
    int tid = threadIdx.x;
    int i = tid >> 4, j = tid & 15;
    float aij = A[i * D_STATE + j];

    __shared__ float hbuf[2][D_STATE];
    if (tid < D_STATE) hbuf[0][tid] = 0.0f;
    __syncthreads();

    int p = 0;
    long mbase = (long)b * SEQ;
    for (int t = 0; t < SEQ; ++t) {
        long m = mbase + t;
        float dtv = dtb[m * D_STATE + i];
        float val = __expf(dtv * aij) * hbuf[p][j];
        #pragma unroll
        for (int o = 8; o >= 1; o >>= 1)
            val += __shfl_xor(val, o, 16);
        if (j == 0) {
            float hn = val + bxb[m * D_STATE + i];
            hbuf[p ^ 1][i] = hn;
            hh[m * 32 + i]            = (_Float16)hn;
            hh[m * 32 + D_STATE + i]  = (_Float16)0.0f;
        }
        __syncthreads();
        p ^= 1;
    }
}

// ---------------------------------------------------------------------------
// y = h @ C^T + Dp * x, stored f16. One wave per 16-row tile, 64 N-tiles,
// single K-step WMMA (K padded to 32 with zeros).
// ---------------------------------------------------------------------------
__global__ __launch_bounds__(256)
void ssm_ykernel(const _Float16* __restrict__ hh, const _Float16* __restrict__ Ch,
                 const float* __restrict__ x, const float* __restrict__ Dp,
                 _Float16* __restrict__ yh) {
    int wave = threadIdx.x >> 5, lane = threadIdx.x & 31;
    long m0 = (long)blockIdx.x * 128 + wave * 16;
    int l15 = lane & 15, h = lane >> 4;

    FragH a;
    const char* hrow = (const char*)hh + (m0 + l15) * 64;
    a.q[0] = *(const uint4*)(hrow + 16 * h);
    a.q[1] = *(const uint4*)(hrow + 32 + 16 * h);

    for (int ntile = 0; ntile < D_MODEL / 16; ++ntile) {
        int d0 = ntile * 16;
        FragH bf;
        const char* crow = (const char*)Ch + (size_t)(d0 + l15) * 64;
        bf.q[0] = *(const uint4*)(crow + 16 * h);
        bf.q[1] = *(const uint4*)(crow + 32 + 16 * h);
        v8f c = {};
        c = wmma_f16(a.v, bf.v, c);

        int d = d0 + l15;
        float dpv = Dp[d];
        #pragma unroll
        for (int r = 0; r < 8; ++r) {
            long m = m0 + r + 8 * h;
            float yv = c[r] + dpv * x[m * D_MODEL + d];
            yh[m * D_MODEL + d] = (_Float16)yv;
        }
    }
}

// ---------------------------------------------------------------------------
// Fused final GEMM: out = [y | x] @ [W_out | W_skip]^T + (b_out + b_skip)
// Block tile 64(M) x 128(N); double-buffered LDS (80B row pitch) staged with
// GLOBAL_LOAD_ASYNC_TO_LDS_B128 when available (ASYNCcnt), else ds stores.
// 8 waves (4x2), each wave = 16x64 via 4 WMMA tiles; K = 2048 in steps of 32.
// ---------------------------------------------------------------------------
#define A_BYTES (64 * 80)
#define B_BYTES (128 * 80)
#define HALF_BYTES (A_BYTES + B_BYTES)

__global__ __launch_bounds__(256)
void ssm_final_gemm(const _Float16* __restrict__ yh, const float* __restrict__ x,
                    const _Float16* __restrict__ Wcomb,
                    const float* __restrict__ b_out, const float* __restrict__ b_skip,
                    float* __restrict__ out) {
    __shared__ __align__(16) unsigned char lds[2 * HALF_BYTES];

    int tid = threadIdx.x;
    long blk = blockIdx.x;
    long mt = blk & 511;                     // 512 M-tiles
    long nt = blk >> 9;                      // 8 N-tiles
    long m0 = mt * 64;
    int  n0 = (int)nt * 128;

    int wave = tid >> 5, lane = tid & 31;
    int wm = wave & 3, wn = wave >> 2;
    int l15 = lane & 15, h = lane >> 4;

    v8f acc[4] = {{}, {}, {}, {}};

    // stage one 64x32 A tile + 128x32 B tile into an LDS half
    auto stage = [&](int ksx, unsigned char* Ab, unsigned char* Bb) {
        int k0 = ksx * 32;
        // ---- A tile ----
        {
            int ra = tid >> 2, cc = tid & 3;
            unsigned char* dst = Ab + ra * 80 + cc * 16;
            if (k0 < D_MODEL) {
                const char* src = (const char*)yh +
                    ((m0 + ra) * D_MODEL + k0) * 2 + cc * 16;
#if HAVE_ASYNC_LDS
                async_copy16(src, dst);
#else
                *(uint4*)dst = *(const uint4*)src;
#endif
            } else {
                const float* src = x + (m0 + ra) * D_MODEL + (k0 - D_MODEL) + cc * 8;
                float4 f0 = ((const float4*)src)[0];
                float4 f1 = ((const float4*)src)[1];
                union { uint4 q; _Float16 e[8]; } cv;
                cv.e[0]=(_Float16)f0.x; cv.e[1]=(_Float16)f0.y;
                cv.e[2]=(_Float16)f0.z; cv.e[3]=(_Float16)f0.w;
                cv.e[4]=(_Float16)f1.x; cv.e[5]=(_Float16)f1.y;
                cv.e[6]=(_Float16)f1.z; cv.e[7]=(_Float16)f1.w;
                *(uint4*)dst = cv.q;
            }
        }
        // ---- B tile: 512 x 16B chunks ----
        #pragma unroll
        for (int u = 0; u < 2; ++u) {
            int idx = tid + u * 256;
            int rb = idx >> 2, cc = idx & 3;
            const char* src = (const char*)Wcomb +
                ((size_t)(n0 + rb) * KCOMB + k0) * 2 + cc * 16;
            unsigned char* dst = Bb + rb * 80 + cc * 16;
#if HAVE_ASYNC_LDS
            async_copy16(src, dst);
#else
            if (u == 0 && (tid & 63) == 0 && ksx + 1 < KCOMB / 32)
                __builtin_prefetch(src + 64, 0, 1);
            *(uint4*)dst = *(const uint4*)src;
#endif
        }
    };

    stage(0, lds, lds + A_BYTES);
    ASYNC_WAIT();
    __syncthreads();

    int buf = 0;
    for (int ks = 0; ks < KCOMB / 32; ++ks) {
        // double-buffer pointers computed arithmetically (no LDS pointer arrays)
        unsigned char* Acur = lds + buf * HALF_BYTES;
        unsigned char* Bcur = Acur + A_BYTES;
        unsigned char* Anxt = lds + (buf ^ 1) * HALF_BYTES;
        unsigned char* Bnxt = Anxt + A_BYTES;

        if (ks + 1 < KCOMB / 32)
            stage(ks + 1, Anxt, Bnxt);             // overlap with WMMAs

        // ---- consume current buffer ----
        FragH a;
        const unsigned char* arow = Acur + (wm * 16 + l15) * 80;
        a.q[0] = *(const uint4*)(arow + 16 * h);
        a.q[1] = *(const uint4*)(arow + 32 + 16 * h);
        #pragma unroll
        for (int t = 0; t < 4; ++t) {
            FragH bf;
            const unsigned char* brow = Bcur + (wn * 64 + t * 16 + l15) * 80;
            bf.q[0] = *(const uint4*)(brow + 16 * h);
            bf.q[1] = *(const uint4*)(brow + 32 + 16 * h);
            acc[t] = wmma_f16(a.v, bf.v, acc[t]);
        }

        ASYNC_WAIT();          // own async copies for buf^1 done
        __syncthreads();       // everyone done reading buf + copies visible
        buf ^= 1;
    }

    // ---- epilogue: + (b_out + b_skip) ----
    #pragma unroll
    for (int t = 0; t < 4; ++t) {
        int n = n0 + wn * 64 + t * 16 + l15;
        float bias = b_out[n] + b_skip[n];
        #pragma unroll
        for (int r = 0; r < 8; ++r) {
            long m = m0 + wm * 16 + r + 8 * h;
            out[m * D_MODEL + n] = acc[t][r] + bias;
        }
    }
}

// ---------------------------------------------------------------------------
extern "C" void kernel_launch(void* const* d_in, const int* in_sizes, int n_in,
                              void* d_out, int out_size, void* d_ws, size_t ws_size,
                              hipStream_t stream) {
    const float* x      = (const float*)d_in[0];
    const float* W_skip = (const float*)d_in[1];
    const float* b_skip = (const float*)d_in[2];
    const float* W_x    = (const float*)d_in[3];
    const float* b_x    = (const float*)d_in[4];
    const float* W_dt   = (const float*)d_in[5];
    const float* b_dt   = (const float*)d_in[6];
    const float* W_out  = (const float*)d_in[7];
    const float* b_out  = (const float*)d_in[8];
    const float* A      = (const float*)d_in[9];
    const float* Bp     = (const float*)d_in[10];
    const float* C      = (const float*)d_in[11];
    const float* Dp     = (const float*)d_in[12];
    float* out = (float*)d_out;

    // workspace layout (bytes)
    char* ws = (char*)d_ws;
    size_t off = 0;
    _Float16* yh    = (_Float16*)(ws + off); off += (size_t)MROWS * D_MODEL * 2;  // 64 MB
    _Float16* Wcomb = (_Float16*)(ws + off); off += (size_t)D_MODEL * KCOMB * 2;  // 4 MB
    _Float16* Wp    = (_Float16*)(ws + off); off += (size_t)32 * D_MODEL * 2;
    _Float16* Ch    = (_Float16*)(ws + off); off += (size_t)D_MODEL * 32 * 2;
    float*    dtb   = (float*)(ws + off);    off += (size_t)MROWS * D_STATE * 4;  // 2 MB
    float*    bxb   = (float*)(ws + off);    off += (size_t)MROWS * D_STATE * 4;  // 2 MB
    _Float16* hh    = (_Float16*)(ws + off); off += (size_t)MROWS * 32 * 2;       // 2 MB

    ssm_prep_wcomb<<<dim3((D_MODEL * KCOMB) / 256), dim3(256), 0, stream>>>(
        W_out, W_skip, Wcomb);
    ssm_prep_small<<<dim3(128), dim3(256), 0, stream>>>(W_dt, W_x, C, Wp, Ch);
    ssm_proj<<<dim3(MROWS / 16 / 8), dim3(256), 0, stream>>>(
        x, Wp, b_dt, b_x, Bp, dtb, bxb);
    ssm_scan<<<dim3(BATCH), dim3(256), 0, stream>>>(dtb, bxb, A, hh);
    ssm_ykernel<<<dim3(MROWS / 128), dim3(256), 0, stream>>>(hh, Ch, x, Dp, yh);
    ssm_final_gemm<<<dim3((MROWS / 64) * (D_MODEL / 128)), dim3(256), 0, stream>>>(
        yh, x, Wcomb, b_out, b_skip, out);
}
